// SwinTransformer_Custom_63024350101836
// MI455X (gfx1250) — matrix operand order repaired and checked by reading, not compiled, not run on hardware
//
#include <hip/hip_runtime.h>
#include <hip/hip_bf16.h>
#include <math.h>

typedef __attribute__((ext_vector_type(16))) _Float16 v16h;
typedef __attribute__((ext_vector_type(8)))  _Float16 v8h;
typedef __attribute__((ext_vector_type(8)))  float    v8f;

// ---------------------------------------------------------------------------
// WMMA fragment loaders from LDS (CDNA5 16-bit A 16x32 / B 32x16 layouts).
// ---------------------------------------------------------------------------
__device__ __forceinline__ v16h frag_a_lds(const _Float16* base, int row0, int stride, int kBase){
  int lane = threadIdx.x & 31;
  int m = row0 + (lane & 15);
  int g = lane >> 4;
  const _Float16* p = base + m * stride + kBase + g * 8;
  v8h lo = *(const v8h*)(p);
  v8h hi = *(const v8h*)(p + 16);
  v16h r;
  #pragma unroll
  for (int i = 0; i < 8; i++){ r[i] = lo[i]; r[i + 8] = hi[i]; }
  return r;
}

__device__ __forceinline__ v16h frag_b_lds(const _Float16* base, int col0, int stride, int kBase){
  int lane = threadIdx.x & 31;
  int n = col0 + (lane & 15);
  int g = lane >> 4;
  const _Float16* p = base + n * stride + kBase + g * 8;
  v8h lo = *(const v8h*)(p);
  v8h hi = *(const v8h*)(p + 16);
  v16h r;
  #pragma unroll
  for (int i = 0; i < 8; i++){ r[i] = lo[i]; r[i + 8] = hi[i]; }
  return r;
}

__device__ __forceinline__ void cvt16(const float4& f0, const float4& f1,
                                      const float4& f2, const float4& f3,
                                      v8h& h0, v8h& h1, float scale){
  h0[0]=(_Float16)(f0.x*scale); h0[1]=(_Float16)(f0.y*scale);
  h0[2]=(_Float16)(f0.z*scale); h0[3]=(_Float16)(f0.w*scale);
  h0[4]=(_Float16)(f1.x*scale); h0[5]=(_Float16)(f1.y*scale);
  h0[6]=(_Float16)(f1.z*scale); h0[7]=(_Float16)(f1.w*scale);
  h1[0]=(_Float16)(f2.x*scale); h1[1]=(_Float16)(f2.y*scale);
  h1[2]=(_Float16)(f2.z*scale); h1[3]=(_Float16)(f2.w*scale);
  h1[4]=(_Float16)(f3.x*scale); h1[5]=(_Float16)(f3.y*scale);
  h1[6]=(_Float16)(f3.z*scale); h1[7]=(_Float16)(f3.w*scale);
}

// ---------------------------------------------------------------------------
// Weight prep: W[K][N] fp32  ->  Wt[N][K] f16 (one-time per launch, small).
// ---------------------------------------------------------------------------
__global__ void wprep_k(const float* __restrict__ W, _Float16* __restrict__ Wt,
                        int K, int N, int total)
{
  int idx = blockIdx.x * blockDim.x + threadIdx.x;
  if (idx >= total) return;
  int n = idx / K, k = idx % K;
  Wt[idx] = (_Float16)W[(long)k * N + n];
}

// ---------------------------------------------------------------------------
// Generic tiled GEMM: C[M,N] = act(A[M,K] @ B[K,N] + bias) (+ residual)
// A fp32 [M][K]; B pre-transposed f16 [N][K]. Block = 128 thr (4 waves),
// tile 64x64, K step 32. REQUIRES K % 32 == 0 (all Swin shapes). OOB rows/
// cols are address-clamped at load and masked at store.
// ---------------------------------------------------------------------------
__global__ __launch_bounds__(128) void gemm_wmma_k(
    const float* __restrict__ A, const _Float16* __restrict__ Bt,
    const float* __restrict__ bias, const float* __restrict__ res,
    float* __restrict__ C, int M, int N, int K, int act)
{
  __shared__ _Float16 As[64 * 32];   // [m][k]
  __shared__ _Float16 Bs[64 * 32];   // [n][k]
  int tid = threadIdx.x, lane = tid & 31, wave = tid >> 5;
  int mBlk = blockIdx.y * 64, nBlk = blockIdx.x * 64;
  int wm = (wave >> 1) * 32, wn = (wave & 1) * 32;

  // per-thread staging coordinates: row/col = tid>>1, k-half = (tid&1)*16
  int rA = tid >> 1, kh = (tid & 1) << 4;
  int gmL = mBlk + rA; if (gmL >= M) gmL = M - 1;   // clamped
  int gnL = nBlk + rA; if (gnL >= N) gnL = N - 1;   // clamped
  const float*    aRow = A  + (long)gmL * K + kh;
  const _Float16* bRow = Bt + (long)gnL * K + kh;

  v8f acc00 = {}, acc01 = {}, acc10 = {}, acc11 = {};

  for (int k0 = 0; k0 < K; k0 += 32){
    // ---- A tile: 4x float4 contiguous loads -> 2x v8h LDS stores ----
    {
      const float* ap = aRow + k0;
      float4 f0 = ((const float4*)ap)[0];
      float4 f1 = ((const float4*)ap)[1];
      float4 f2 = ((const float4*)ap)[2];
      float4 f3 = ((const float4*)ap)[3];
      if (k0 + 32 < K){
        __builtin_prefetch(ap + 32, 0, 1);
        __builtin_prefetch(bRow + k0 + 32, 0, 1);
      }
      v8h h0, h1; cvt16(f0, f1, f2, f3, h0, h1, 1.f);
      *(v8h*)&As[rA * 32 + kh]     = h0;
      *(v8h*)&As[rA * 32 + kh + 8] = h1;
    }
    // ---- B tile: 2x b128 contiguous f16 loads -> 2x v8h LDS stores ----
    {
      v8h h0 = *(const v8h*)(bRow + k0);
      v8h h1 = *(const v8h*)(bRow + k0 + 8);
      *(v8h*)&Bs[rA * 32 + kh]     = h0;
      *(v8h*)&Bs[rA * 32 + kh + 8] = h1;
    }
    __syncthreads();
    v16h a0 = frag_a_lds(As, wm,      32, 0);
    v16h a1 = frag_a_lds(As, wm + 16, 32, 0);
    v16h b0 = frag_b_lds(Bs, wn,      32, 0);
    v16h b1 = frag_b_lds(Bs, wn + 16, 32, 0);
    acc00 = __builtin_amdgcn_wmma_f32_16x16x32_f16(false, a0, false, b0, (short)0, acc00, false, false);
    acc01 = __builtin_amdgcn_wmma_f32_16x16x32_f16(false, a0, false, b1, (short)0, acc01, false, false);
    acc10 = __builtin_amdgcn_wmma_f32_16x16x32_f16(false, a1, false, b0, (short)0, acc10, false, false);
    acc11 = __builtin_amdgcn_wmma_f32_16x16x32_f16(false, a1, false, b1, (short)0, acc11, false, false);
    __syncthreads();
  }

  int nLane = lane & 15, mHalf = (lane >> 4) << 3;
  #pragma unroll
  for (int i = 0; i < 2; i++){
    #pragma unroll
    for (int j = 0; j < 2; j++){
      v8f acc = (i == 0) ? ((j == 0) ? acc00 : acc01) : ((j == 0) ? acc10 : acc11);
      int gn  = nBlk + wn + j * 16 + nLane;
      int gmB = mBlk + wm + i * 16 + mHalf;
      if (gn < N){
        float bv = bias ? bias[gn] : 0.f;
        #pragma unroll
        for (int r = 0; r < 8; r++){
          int gm = gmB + r;
          if (gm < M){
            float v = acc[r] + bv;
            if (act == 1) v = 0.5f * v * (1.f + erff(v * 0.70710678118654752f));
            if (res)  v += res[(long)gm * N + gn];
            C[(long)gm * N + gn] = v;
          }
        }
      }
    }
  }
}

// ---------------------------------------------------------------------------
// Windowed attention: one block per (window, head). 49 tokens, head_dim 32.
// Padded to 64x64; WMMA scores + in-LDS softmax (+rpb, +shift mask) + WMMA AV.
// ---------------------------------------------------------------------------
__device__ __forceinline__ int zone3(int x, int S){
  return (x < S - 7) ? 0 : ((x < S - 3) ? 1 : 2);
}

__global__ __launch_bounds__(128) void win_attn_k(
    const float* __restrict__ qkv, const float* __restrict__ rpb,
    float* __restrict__ o, int C, int nHeads, int nw, int H, int shift)
{
  __shared__ _Float16 Qs[64 * 32];  // [m][d]
  __shared__ _Float16 Ks[64 * 32];  // [n][d]
  __shared__ _Float16 Vt[32 * 64];  // [d][t]
  __shared__ float    Sc[64 * 64];
  __shared__ _Float16 At[64 * 64];  // [m][t]

  int tid = threadIdx.x, lane = tid & 31, wave = tid >> 5;
  int win = blockIdx.x, head = blockIdx.y;
  long rs = 3L * C;
  const float* qb = qkv + (long)win * 49 * rs + (long)head * 32;
  const float* kb = qb + C;
  const float* vb = qb + 2L * C;
  const float scale = 0.17677669529663687f; // 1/sqrt(32)

  // ---- Q/K staging: thread owns (row = tid>>1, d-half = (tid&1)*16) ----
  {
    int m = tid >> 1, dh = (tid & 1) << 4;
    int mc = (m < 49) ? m : 48;                 // clamped row (finite data)
    const float* qp = qb + (long)mc * rs + dh;
    const float* kp = kb + (long)mc * rs + dh;
    float4 q0 = ((const float4*)qp)[0], q1 = ((const float4*)qp)[1],
           q2 = ((const float4*)qp)[2], q3 = ((const float4*)qp)[3];
    float4 k0 = ((const float4*)kp)[0], k1 = ((const float4*)kp)[1],
           k2 = ((const float4*)kp)[2], k3 = ((const float4*)kp)[3];
    v8h h0, h1;
    cvt16(q0, q1, q2, q3, h0, h1, scale);
    *(v8h*)&Qs[m * 32 + dh] = h0; *(v8h*)&Qs[m * 32 + dh + 8] = h1;
    cvt16(k0, k1, k2, k3, h0, h1, 1.f);
    *(v8h*)&Ks[m * 32 + dh] = h0; *(v8h*)&Ks[m * 32 + dh + 8] = h1;
  }
  // ---- V staging (transposed): thread owns (d = tid>>2, t-half = (tid&3)*16)
  {
    int d = tid >> 2, th = (tid & 3) << 4;
    v8h h0, h1;
    #pragma unroll
    for (int j = 0; j < 8; j++){
      int t0 = th + j;      int tc0 = (t0 < 49) ? t0 : 48;
      h0[j] = (_Float16)vb[(long)tc0 * rs + d];
    }
    #pragma unroll
    for (int j = 0; j < 8; j++){
      int t1 = th + 8 + j;  int tc1 = (t1 < 49) ? t1 : 48;
      h1[j] = (_Float16)vb[(long)tc1 * rs + d];
    }
    *(v8h*)&Vt[d * 64 + th] = h0; *(v8h*)&Vt[d * 64 + th + 8] = h1;
  }
  for (int idx = tid; idx < 4096; idx += 128) At[idx] = (_Float16)0.f;
  __syncthreads();

  // --- scores: each wave does rows [16*wave, 16*wave+16) x all 64 cols ---
  {
    v16h a = frag_a_lds(Qs, wave * 16, 32, 0);
    int nLane = lane & 15, mHalf = (lane >> 4) << 3;
    #pragma unroll
    for (int j = 0; j < 4; j++){
      v16h b = frag_b_lds(Ks, j * 16, 32, 0);
      v8f acc = {};
      acc = __builtin_amdgcn_wmma_f32_16x16x32_f16(false, a, false, b, (short)0, acc, false, false);
      int n = j * 16 + nLane, mB = wave * 16 + mHalf;
      #pragma unroll
      for (int r = 0; r < 8; r++) Sc[(mB + r) * 64 + n] = acc[r];
    }
  }
  __syncthreads();

  // --- softmax per row (rel pos bias + shifted-window mask on the fly) ---
  int winLocal = win % (nw * nw);
  int wh = winLocal / nw, wwi = winLocal % nw;
  for (int m = tid; m < 49; m += 128){
    int ih = m / 7, iw = m % 7;
    int labI = 0;
    if (shift) labI = zone3(wh * 7 + ih, H) * 3 + zone3(wwi * 7 + iw, H);
    float mx = -1e30f;
    for (int n = 0; n < 49; n++){
      int jh = n / 7, jw = n % 7;
      int rel = (ih - jh + 6) * 13 + (iw - jw + 6);
      float sv = Sc[m * 64 + n] + rpb[rel * nHeads + head];
      if (shift){
        int labJ = zone3(wh * 7 + jh, H) * 3 + zone3(wwi * 7 + jw, H);
        if (labJ != labI) sv -= 100.f;
      }
      Sc[m * 64 + n] = sv;
      mx = fmaxf(mx, sv);
    }
    float sum = 0.f;
    for (int n = 0; n < 49; n++){ float e = __expf(Sc[m * 64 + n] - mx); Sc[m * 64 + n] = e; sum += e; }
    float inv = 1.f / sum;
    for (int n = 0; n < 49; n++) At[m * 64 + n] = (_Float16)(Sc[m * 64 + n] * inv);
  }
  __syncthreads();

  // --- AV: rows [16*wave..), cols 0..31, K=64 (2 WMMA steps) ---
  {
    int nLane = lane & 15, mHalf = (lane >> 4) << 3;
    #pragma unroll
    for (int j = 0; j < 2; j++){
      v8f acc = {};
      #pragma unroll
      for (int kt = 0; kt < 2; kt++){
        v16h a = frag_a_lds(At, wave * 16, 64, kt * 32);
        v16h b = frag_b_lds(Vt, j * 16,   64, kt * 32);
        acc = __builtin_amdgcn_wmma_f32_16x16x32_f16(false, a, false, b, (short)0, acc, false, false);
      }
      int d = j * 16 + nLane, mB = wave * 16 + mHalf;
      #pragma unroll
      for (int r = 0; r < 8; r++){
        int m = mB + r;
        if (m < 49) o[((long)win * 49 + m) * C + head * 32 + d] = acc[r];
      }
    }
  }
}

// ---------------------------------------------------------------------------
// LayerNorm: one wave32 per row (4 rows / 128-thread block).
// ---------------------------------------------------------------------------
__global__ __launch_bounds__(128) void layernorm_k(
    const float* __restrict__ x, const float* __restrict__ g,
    const float* __restrict__ b, float* __restrict__ o, int T, int C)
{
  int wave = threadIdx.x >> 5, lane = threadIdx.x & 31;
  int row = blockIdx.x * 4 + wave;
  if (row >= T) return;
  const float* xr = x + (long)row * C;
  float s = 0.f;
  for (int c = lane; c < C; c += 32) s += xr[c];
  #pragma unroll
  for (int t = 16; t; t >>= 1) s += __shfl_xor(s, t, 32);
  float mu = s / (float)C;
  float v = 0.f;
  for (int c = lane; c < C; c += 32){ float d = xr[c] - mu; v += d * d; }
  #pragma unroll
  for (int t = 16; t; t >>= 1) v += __shfl_xor(v, t, 32);
  float inv = rsqrtf(v / (float)C + 1e-5f);
  float* orow = o + (long)row * C;
  for (int c = lane; c < C; c += 32) orow[c] = (xr[c] - mu) * inv * g[c] + b[c];
}

// ---------------------------------------------------------------------------
// Patch embed: 4x4 stride-4 conv (K=48) + bias. y: [16,56,56,96] NHWC.
// ---------------------------------------------------------------------------
__global__ void patch_embed_k(const float* __restrict__ x, const float* __restrict__ w,
                              const float* __restrict__ bias, float* __restrict__ y, int total)
{
  int idx = blockIdx.x * blockDim.x + threadIdx.x;
  if (idx >= total) return;
  int c = idx % 96; int t = idx / 96;
  int wp = t % 56; int hp = (t / 56) % 56; int b = t / (56 * 56);
  float s = bias[c];
  #pragma unroll
  for (int i = 0; i < 4; i++)
    #pragma unroll
    for (int j = 0; j < 4; j++)
      #pragma unroll
      for (int ci = 0; ci < 3; ci++)
        s += x[(((long)b * 3 + ci) * 224 + hp * 4 + i) * 224 + wp * 4 + j]
           * w[((i * 4 + j) * 3 + ci) * 96 + c];
  y[idx] = s;
}

// --------------------- window partition / unpartition ----------------------
__global__ void win_part_k(const float* __restrict__ in, float* __restrict__ out,
                           int H, int W, int C, int shift, int total)
{
  int idx = blockIdx.x * blockDim.x + threadIdx.x;
  if (idx >= total) return;
  int c = idx % C; int t = idx / C;
  int i = t % 49; int win = t / 49;
  int nw = W / 7;
  int ww = win % nw; int rest = win / nw;
  int wh = rest % (H / 7); int b = rest / (H / 7);
  int hh = wh * 7 + i / 7, wc = ww * 7 + i % 7;
  int hs = (hh + shift) % H, ws = (wc + shift) % W;
  out[idx] = in[(((long)b * H + hs) * W + ws) * C + c];
}

__global__ void win_unpart_add_k(const float* __restrict__ proj, float* __restrict__ y,
                                 int H, int W, int C, int shift, int total)
{
  int idx = blockIdx.x * blockDim.x + threadIdx.x;
  if (idx >= total) return;
  int c = idx % C; int t = idx / C;
  int w = t % W; int h = (t / W) % H; int b = t / (W * H);
  int hp = (h - shift + H) % H, wp = (w - shift + W) % W;
  int wh = hp / 7, ihh = hp % 7, ww = wp / 7, iww = wp % 7;
  int nw = W / 7;
  int win = ((long)b * (H / 7) + wh) * nw + ww;
  int i = ihh * 7 + iww;
  y[idx] += proj[((long)win * 49 + i) * C + c];
}

// ------------------------------- patch merge -------------------------------
__global__ void merge_gather_k(const float* __restrict__ y, float* __restrict__ out,
                               int H, int W, int C, int total)
{
  int idx = blockIdx.x * blockDim.x + threadIdx.x;
  if (idx >= total) return;
  int C4 = 4 * C;
  int c4 = idx % C4; int t = idx / C4;
  int W2 = W / 2;
  int w2 = t % W2; int h2 = (t / W2) % (H / 2); int b = t / (W2 * (H / 2));
  int g = c4 / C, c = c4 % C;
  int dh = (g == 1 || g == 3) ? 1 : 0;
  int dw = (g >= 2) ? 1 : 0;
  out[idx] = y[(((long)b * H + 2 * h2 + dh) * W + 2 * w2 + dw) * C + c];
}

// ---------------------------- final pooling/head ---------------------------
__global__ void mean_k(const float* __restrict__ x, float* __restrict__ f, int T, int C)
{
  int idx = blockIdx.x * blockDim.x + threadIdx.x;
  if (idx >= 16 * C) return;
  int b = idx / C, c = idx % C;
  float s = 0.f;
  for (int t = 0; t < T; t++) s += x[((long)b * T + t) * C + c];
  f[idx] = s / (float)T;
}

__global__ void head_k(const float* __restrict__ f, const float* __restrict__ w,
                       const float* __restrict__ b, float* __restrict__ logits)
{
  int tid = threadIdx.x;
  if (tid >= 32) return;
  int bb = tid >> 1, n = tid & 1;
  float s = b[n];
  for (int k = 0; k < 768; k++) s += f[bb * 768 + k] * w[k * 2 + n];
  logits[bb * 2 + n] = s;
}

// ---------------------------------------------------------------------------
// Host orchestration
// ---------------------------------------------------------------------------
extern "C" void kernel_launch(void* const* d_in, const int* in_sizes, int n_in,
                              void* d_out, int out_size, void* d_ws, size_t ws_size,
                              hipStream_t stream)
{
  (void)in_sizes; (void)n_in; (void)out_size; (void)ws_size;
  const int DEPTHS[4] = {2, 2, 6, 2};
  const int HEADS[4]  = {3, 6, 12, 24};

  const float* x = (const float*)d_in[0];
  int pi = 1;
  // blocks: dict keys sorted: fc1_b fc1_w fc2_b fc2_w n1_b n1_g n2_b n2_g
  //                           proj_b proj_w qkv_b qkv_w rpb
  const float* blk[4][6][13];
  for (int s = 0; s < 4; s++)
    for (int i = 0; i < DEPTHS[s]; i++)
      for (int k = 0; k < 13; k++) blk[s][i][k] = (const float*)d_in[pi++];
  const float* head_b = (const float*)d_in[pi++];
  const float* head_w = (const float*)d_in[pi++];
  const float* mrg[3][3]; // n_b, n_g, red_w
  for (int s = 0; s < 3; s++)
    for (int k = 0; k < 3; k++) mrg[s][k] = (const float*)d_in[pi++];
  const float* norm_b  = (const float*)d_in[pi++];
  const float* norm_g  = (const float*)d_in[pi++];
  const float* patch_b = (const float*)d_in[pi++];
  const float* patch_w = (const float*)d_in[pi++];
  const float* pe_n_b  = (const float*)d_in[pi++];
  const float* pe_n_g  = (const float*)d_in[pi++];

  // ---- workspace carve-out (fp32 buffers, then f16 weight cache) ----
  float* wsf = (float*)d_ws;
  size_t off = 0;
  const size_t TOK0 = 16 * 56 * 56; // 50176
  float* bufY   = wsf + off; off += TOK0 * 96;   // activations
  float* bufN   = wsf + off; off += TOK0 * 96;   // layernorm output
  float* bufX   = wsf + off; off += TOK0 * 96;   // windowed / merge-gather
  float* bufO   = wsf + off; off += TOK0 * 96;   // attention output
  float* bufP   = wsf + off; off += TOK0 * 96;   // proj output
  float* bufBig = wsf + off; off += TOK0 * 384;  // qkv / mlp hidden

  _Float16* wsh = (_Float16*)(wsf + off);
  size_t hoff = 0;
  auto eb = [](long n){ return (unsigned)((n + 255) / 256); };
  auto prep = [&](const float* W, int K, int N) -> const _Float16* {
    _Float16* t = wsh + hoff; hoff += (size_t)N * K;
    wprep_k<<<eb((long)N * K), 256, 0, stream>>>(W, t, K, N, N * K);
    return t;
  };

  // ---- one-time per-launch f16 weight transposes ----
  const _Float16 *qkvW[4][6], *projW[4][6], *fc1W[4][6], *fc2W[4][6], *redW[3];
  for (int s = 0; s < 4; s++){
    int C = 96 << s;
    for (int i = 0; i < DEPTHS[s]; i++){
      const float* const* P = blk[s][i];
      qkvW[s][i]  = prep(P[11], C,     3 * C);
      projW[s][i] = prep(P[9],  C,     C);
      fc1W[s][i]  = prep(P[1],  C,     4 * C);
      fc2W[s][i]  = prep(P[3],  4 * C, C);
    }
    if (s < 3) redW[s] = prep(mrg[s][2], 4 * C, 2 * C);
  }

  int Hc = 56, Cc = 96;
  int TOK = 16 * Hc * Hc;

  // patch embed + embedding layernorm (in-place)
  patch_embed_k<<<eb((long)TOK * Cc), 256, 0, stream>>>(x, patch_w, patch_b, bufY, TOK * Cc);
  layernorm_k<<<(TOK + 3) / 4, 128, 0, stream>>>(bufY, pe_n_g, pe_n_b, bufY, TOK, Cc);

  for (int s = 0; s < 4; s++){
    int nHd = HEADS[s];
    int nw = Hc / 7, nW = 16 * nw * nw;
    TOK = 16 * Hc * Hc;
    for (int i = 0; i < DEPTHS[s]; i++){
      const float* const* P = blk[s][i];
      int shift = ((i % 2) == 1 && 7 < Hc) ? 3 : 0;
      // ---- attention branch ----
      layernorm_k<<<(TOK + 3) / 4, 128, 0, stream>>>(bufY, P[5], P[4], bufN, TOK, Cc);
      win_part_k<<<eb((long)TOK * Cc), 256, 0, stream>>>(bufN, bufX, Hc, Hc, Cc, shift, TOK * Cc);
      gemm_wmma_k<<<dim3((3 * Cc + 63) / 64, (TOK + 63) / 64), 128, 0, stream>>>(
          bufX, qkvW[s][i], P[10], nullptr, bufBig, TOK, 3 * Cc, Cc, 0);
      win_attn_k<<<dim3(nW, nHd), 128, 0, stream>>>(bufBig, P[12], bufO, Cc, nHd, nw, Hc, shift);
      gemm_wmma_k<<<dim3((Cc + 63) / 64, (TOK + 63) / 64), 128, 0, stream>>>(
          bufO, projW[s][i], P[8], nullptr, bufP, TOK, Cc, Cc, 0);
      win_unpart_add_k<<<eb((long)TOK * Cc), 256, 0, stream>>>(bufP, bufY, Hc, Hc, Cc, shift, TOK * Cc);
      // ---- MLP branch ----
      layernorm_k<<<(TOK + 3) / 4, 128, 0, stream>>>(bufY, P[7], P[6], bufN, TOK, Cc);
      gemm_wmma_k<<<dim3((4 * Cc + 63) / 64, (TOK + 63) / 64), 128, 0, stream>>>(
          bufN, fc1W[s][i], P[0], nullptr, bufBig, TOK, 4 * Cc, Cc, 1);
      gemm_wmma_k<<<dim3((Cc + 63) / 64, (TOK + 63) / 64), 128, 0, stream>>>(
          bufBig, fc2W[s][i], P[2], bufY, bufY, TOK, Cc, 4 * Cc, 0);
    }
    if (s < 3){
      int T2 = TOK / 4;
      merge_gather_k<<<eb((long)T2 * 4 * Cc), 256, 0, stream>>>(bufY, bufX, Hc, Hc, Cc, T2 * 4 * Cc);
      layernorm_k<<<(T2 + 3) / 4, 128, 0, stream>>>(bufX, mrg[s][1], mrg[s][0], bufN, T2, 4 * Cc);
      gemm_wmma_k<<<dim3((2 * Cc + 63) / 64, (T2 + 63) / 64), 128, 0, stream>>>(
          bufN, redW[s], nullptr, nullptr, bufY, T2, 2 * Cc, 4 * Cc, 0);
      Hc /= 2; Cc *= 2;
    }
  }

  // final norm -> mean pool -> head. d_out = [logits(32) | f(16*768)]
  int Tf = 16 * Hc * Hc; // 16*49
  layernorm_k<<<(Tf + 3) / 4, 128, 0, stream>>>(bufY, norm_g, norm_b, bufN, Tf, Cc);
  float* outp = (float*)d_out;
  mean_k<<<eb(16 * 768), 256, 0, stream>>>(bufN, outp + 32, Hc * Hc, 768);
  head_k<<<1, 32, 0, stream>>>(outp + 32, head_w, head_b, outp);
}
